// MolecularGNN_29935922053239
// MI455X (gfx1250) — matrix operand order repaired and checked by reading, not compiled, not run on hardware
//
#include <hip/hip_runtime.h>
#include <hip/hip_bf16.h>

#define N_NODES 50000
#define N_EDGES 800000
#define NB      256
#define HIDC    64
#define HEADS   4
#define EDGE_F  8
#define GLOBF   4
#define HC      256
#define ET      (N_EDGES + N_NODES)
#define EPSLN   1e-5f

typedef __attribute__((ext_vector_type(16))) _Float16 v16h;
typedef __attribute__((ext_vector_type(8)))  float    v8f;
typedef __attribute__((ext_vector_type(4)))  int      v4i;

// Async global->LDS staging (ASYNCcnt path), guarded so absence cannot break compile.
#if defined(__has_builtin)
#if __has_builtin(__builtin_amdgcn_global_load_async_to_lds_b128)
#define USE_ASYNC_LDS 1
#endif
#endif
#ifndef USE_ASYNC_LDS
#define USE_ASYNC_LDS 0
#endif

#if USE_ASYNC_LDS
typedef __attribute__((address_space(1))) v4i* gv4i_ptr;   // global int4*
typedef __attribute__((address_space(3))) v4i* lv4i_ptr;   // LDS int4*
__device__ __forceinline__ void wait_asynccnt0() {
#if __has_builtin(__builtin_amdgcn_s_wait_asynccnt)
    __builtin_amdgcn_s_wait_asynccnt(0);
#else
    asm volatile("s_wait_asynccnt 0x0" ::: "memory");
#endif
}
#endif

// ---------- helpers ----------
__device__ __forceinline__ unsigned f2mono(float f) {
    unsigned u = __float_as_uint(f);
    return (u & 0x80000000u) ? ~u : (u | 0x80000000u);
}
__device__ __forceinline__ float mono2f(unsigned u) {
    return (u & 0x80000000u) ? __uint_as_float(u & 0x7fffffffu)
                             : __uint_as_float(~u);
}

__global__ void zero_kernel(float* p, long n) {
    long i = (long)blockIdx.x * blockDim.x + threadIdx.x;
    long s = (long)gridDim.x * blockDim.x;
    for (; i < n; i += s) p[i] = 0.f;
}

// per-feature column sums of edge_attr (for PyG fill_value='mean' self loops)
__global__ void edge_colsum(const float* __restrict__ ea, float* sums) {
    float loc[EDGE_F];
#pragma unroll
    for (int f = 0; f < EDGE_F; ++f) loc[f] = 0.f;
    long i = (long)blockIdx.x * blockDim.x + threadIdx.x;
    long s = (long)gridDim.x * blockDim.x;
    for (; i < N_EDGES; i += s) {
        const float* p = ea + i * EDGE_F;
#pragma unroll
        for (int f = 0; f < EDGE_F; ++f) loc[f] += p[f];
    }
#pragma unroll
    for (int f = 0; f < EDGE_F; ++f) atomicAdd(&sums[f], loc[f]);
}

// Mproj[f][h] = sum_c We[f, h*64+c] * Ae[h, c]   (folds e = ea@We, a_e = e.Ae)
// ae_self[h]  = sum_f mean_ea[f] * Mproj[f][h]
__global__ void layer_setup(const float* __restrict__ We, const float* __restrict__ Ae,
                            const float* __restrict__ sums, int H,
                            float* Mproj, float* ae_self) {
    int t = threadIdx.x;               // 64 threads
    if (t < EDGE_F * H) {
        int f = t / H, h = t % H;
        float s = 0.f;
        for (int c = 0; c < HIDC; ++c)
            s += We[f * (H * HIDC) + h * HIDC + c] * Ae[h * HIDC + c];
        Mproj[f * H + h] = s;
    }
    __syncthreads();
    if (t < H) {
        float s = 0.f;
        for (int f = 0; f < EDGE_F; ++f)
            s += (sums[f] / (float)N_EDGES) * Mproj[f * H + t];
        ae_self[t] = s;
    }
}

// ---------- WMMA GEMM: Y[nrows, ncols] = X[nrows, KDIM] @ W[KDIM, ncols] ----------
// Block = 256 thr (8 waves). Wave -> 16-row x 64-col panel (4 wmma tiles).
// W strip (KDIM x 64) staged in LDS pre-swizzled into B-fragment order so a
// B operand is one contiguous 32B read per lane. If the toolchain exposes the
// gfx1250 async-to-LDS path, the raw f32 strip is DMA'd via ASYNCcnt first.
template <int KDIM>
__global__ void gemm_wmma(const float* __restrict__ X, const float* __restrict__ W,
                          float* __restrict__ Y, int nrows, int ncols) {
    __shared__ _Float16 wlds[KDIM * 64];
    const int tid = threadIdx.x;
    const int colbase = blockIdx.y * 64;

#if USE_ASYNC_LDS
    __shared__ float wraw[KDIM * 64];
    // stage raw f32 strip via async global->LDS, 16B per lane
    for (int c = tid; c < KDIM * 16; c += 256) {          // KDIM*16 chunks of 16B
        int k = c >> 4, j = c & 15;                       // strip row k, 4-float chunk j
        const float* g = W + (long)k * ncols + colbase + j * 4;
        float* l = &wraw[k * 64 + j * 4];
        __builtin_amdgcn_global_load_async_to_lds_b128((gv4i_ptr)g, (lv4i_ptr)l, 0, 0);
    }
    wait_asynccnt0();
    __syncthreads();
    // fragment-swizzle LDS->LDS with f32 -> f16 conversion
    for (int e = tid; e < KDIM * 64; e += 256) {
        int k = e >> 6, c = e & 63;
        int kt = k >> 5, kr = k & 31;
        int half = kr >> 4, ki = kr & 15;
        int nt = c >> 4, nn = c & 15;
        int ln = half * 16 + nn;
        wlds[((kt * 4 + nt) * 32 + ln) * 16 + ki] = (_Float16)wraw[k * 64 + c];
    }
#else
    // cooperative load + fragment-swizzle of W strip, f32 -> f16
    for (int e = tid; e < KDIM * 64; e += 256) {
        int k = e >> 6, c = e & 63;
        int kt = k >> 5, kr = k & 31;
        int half = kr >> 4, ki = kr & 15;
        int nt = c >> 4, nn = c & 15;
        int ln = half * 16 + nn;
        wlds[((kt * 4 + nt) * 32 + ln) * 16 + ki] =
            (_Float16)W[(long)k * ncols + colbase + c];
    }
#endif
    __syncthreads();

    const int wave = tid >> 5, lane = tid & 31;
    const int rowbase = blockIdx.x * 128 + wave * 16;
    if (rowbase >= nrows) return;              // nrows % 16 == 0 (50000)

    const int r  = rowbase + (lane & 15);
    const int kg = lane >> 4;                  // K half select for A frag
    const float* xrow = X + (long)r * KDIM;
    const v16h* bw = (const v16h*)wlds;

    v8f acc0 = {}, acc1 = {}, acc2 = {}, acc3 = {};
    for (int kt = 0; kt < KDIM / 32; ++kt) {
        // A fragment: per lane K = kg*16..+15 (contiguous, matches B swizzle)
        v16h a;
        const float* xp = xrow + kt * 32 + kg * 16;
#pragma unroll
        for (int j = 0; j < 8; ++j) a[j]     = (_Float16)xp[j];
#pragma unroll
        for (int j = 0; j < 8; ++j) a[8 + j] = (_Float16)xp[8 + j];

        v16h b;
        b = bw[(kt * 4 + 0) * 32 + lane];
        acc0 = __builtin_amdgcn_wmma_f32_16x16x32_f16(false, a, false, b, (short)0, acc0, false, false);
        b = bw[(kt * 4 + 1) * 32 + lane];
        acc1 = __builtin_amdgcn_wmma_f32_16x16x32_f16(false, a, false, b, (short)0, acc1, false, false);
        b = bw[(kt * 4 + 2) * 32 + lane];
        acc2 = __builtin_amdgcn_wmma_f32_16x16x32_f16(false, a, false, b, (short)0, acc2, false, false);
        b = bw[(kt * 4 + 3) * 32 + lane];
        acc3 = __builtin_amdgcn_wmma_f32_16x16x32_f16(false, a, false, b, (short)0, acc3, false, false);
    }

    // C/D layout: VGPR v -> row rowbase + v + (lane>=16 ? 8 : 0), col = lane&15
    const int colb   = colbase + (lane & 15);
    const int rowoff = rowbase + kg * 8;
#pragma unroll
    for (int v = 0; v < 8; ++v) {
        long rb = (long)(rowoff + v) * ncols + colb;
        Y[rb]      = acc0[v];
        Y[rb + 16] = acc1[v];
        Y[rb + 32] = acc2[v];
        Y[rb + 48] = acc3[v];
    }
}

// a_s[n,h], a_d[n,h] from projected node features
__global__ void node_attn(const float* __restrict__ h, const float* __restrict__ As,
                          const float* __restrict__ Ad, int H,
                          float* __restrict__ a_s, float* __restrict__ a_d) {
    int i = blockIdx.x * blockDim.x + threadIdx.x;
    if (i >= N_NODES * H) return;
    int n = i / H, hh = i % H;
    const float* row = h + (long)n * (H * HIDC) + hh * HIDC;
    float ss = 0.f, sd = 0.f;
    for (int c = 0; c < HIDC; ++c) {
        float v = row[c];
        ss += v * As[hh * HIDC + c];
        sd += v * Ad[hh * HIDC + c];
    }
    a_s[i] = ss;
    a_d[i] = sd;
}

// pass 1: alpha = leaky_relu(a_s[src]+a_d[dst]+a_e), segment max via monotone atomicMax
__global__ void edge_alpha(const int* __restrict__ ei, const float* __restrict__ ea,
                           const float* __restrict__ Mproj, const float* __restrict__ ae_self,
                           const float* __restrict__ a_s, const float* __restrict__ a_d,
                           int H, float* __restrict__ alpha, unsigned* __restrict__ menc) {
    int e = blockIdx.x * blockDim.x + threadIdx.x;
    if (e >= ET) return;
    int src, dst;
    float aev[HEADS];
    if (e < N_EDGES) {
        src = ei[e];
        dst = ei[N_EDGES + e];
        const float* p = ea + (long)e * EDGE_F;
        for (int h = 0; h < H; ++h) {
            float s = 0.f;
#pragma unroll
            for (int f = 0; f < EDGE_F; ++f) s += p[f] * Mproj[f * H + h];
            aev[h] = s;
        }
    } else {
        src = dst = e - N_EDGES;                 // self loop
        for (int h = 0; h < H; ++h) aev[h] = ae_self[h];
    }
    for (int h = 0; h < H; ++h) {
        float x  = a_s[(long)src * H + h] + a_d[(long)dst * H + h] + aev[h];
        float al = x > 0.f ? x : 0.2f * x;
        alpha[(long)e * H + h] = al;
        atomicMax(&menc[(long)dst * H + h], f2mono(al));
    }
}

// pass 2: ex = exp(alpha - m[dst]); den[dst] += ex (overwrite alpha with ex)
__global__ void edge_exp(const int* __restrict__ ei, int H,
                         const unsigned* __restrict__ menc,
                         float* __restrict__ alpha, float* __restrict__ den) {
    int e = blockIdx.x * blockDim.x + threadIdx.x;
    if (e >= ET) return;
    int dst = (e < N_EDGES) ? ei[N_EDGES + e] : e - N_EDGES;
    for (int h = 0; h < H; ++h) {
        float m  = mono2f(menc[(long)dst * H + h]);
        float ex = __expf(alpha[(long)e * H + h] - m);
        alpha[(long)e * H + h] = ex;
        atomicAdd(&den[(long)dst * H + h], ex);
    }
}

// pass 3: out[dst] += (ex/den[dst]) * h[src]; one wave per edge, HCC/32 ch/lane
template <int HCC>
__global__ void edge_agg(const int* __restrict__ ei, const float* __restrict__ h,
                         const float* __restrict__ ex, const float* __restrict__ den,
                         int H, float* __restrict__ out) {
    const int CPL = HCC / 32;
    int wave = threadIdx.x >> 5, lane = threadIdx.x & 31;
    long e = (long)blockIdx.x * 8 + wave;
    if (e >= ET) return;
    int src, dst;
    if (e < N_EDGES) { src = ei[e]; dst = ei[N_EDGES + e]; }
    else             { src = dst = (int)e - N_EDGES; }
    int c0 = lane * CPL;
    int hh = c0 >> 6;                           // head index (HIDC == 64)
    const float* hs = h + (long)src * HCC + c0;
    __builtin_prefetch(hs, 0, 0);               // global_prefetch_b8
    float coef = ex[e * H + hh] / den[(long)dst * H + hh];
    float* op = out + (long)dst * HCC + c0;
#pragma unroll
    for (int j = 0; j < CPL; ++j) atomicAdd(&op[j], coef * hs[j]);
}

// out = relu(LN(agg + b) * g + be), in place; one wave per node
template <int HCC>
__global__ void bias_ln_relu(float* __restrict__ hio, const float* __restrict__ b,
                             const float* __restrict__ g, const float* __restrict__ be) {
    const int CPL = HCC / 32;
    int wave = threadIdx.x >> 5, lane = threadIdx.x & 31;
    long n = (long)blockIdx.x * 8 + wave;
    if (n >= N_NODES) return;
    float* row = hio + n * HCC;
    float v[CPL];
    float s = 0.f;
#pragma unroll
    for (int j = 0; j < CPL; ++j) {
        int c = lane * CPL + j;
        v[j] = row[c] + b[c];
        s += v[j];
    }
    for (int off = 16; off; off >>= 1) s += __shfl_xor(s, off, 32);
    float mu = s / (float)HCC;
    float q = 0.f;
#pragma unroll
    for (int j = 0; j < CPL; ++j) { float d = v[j] - mu; q += d * d; }
    for (int off = 16; off; off >>= 1) q += __shfl_xor(q, off, 32);
    float inv = rsqrtf(q / (float)HCC + EPSLN);
#pragma unroll
    for (int j = 0; j < CPL; ++j) {
        int c = lane * CPL + j;
        float y = (v[j] - mu) * inv * g[c] + be[c];
        row[c] = fmaxf(y, 0.f);
    }
}

// mean-pool accumulation: one wave per node, 2 channels/lane
__global__ void pool_accum(const float* __restrict__ h3, const int* __restrict__ batch,
                           float* pooled, float* cnt) {
    int wave = threadIdx.x >> 5, lane = threadIdx.x & 31;
    long n = (long)blockIdx.x * 8 + wave;
    if (n >= N_NODES) return;
    int b = batch[n];
    atomicAdd(&pooled[(long)b * HIDC + lane * 2],     h3[n * HIDC + lane * 2]);
    atomicAdd(&pooled[(long)b * HIDC + lane * 2 + 1], h3[n * HIDC + lane * 2 + 1]);
    if (lane == 0) atomicAdd(&cnt[b], 1.f);
}

// head MLP: hid = relu([pooled/cnt, glob] @ fcW1 + fcb1); out = hid @ fcW2 + fcb2
__global__ void head_mlp(const float* __restrict__ pooled, const float* __restrict__ cnt,
                         const float* __restrict__ glob,
                         const float* __restrict__ fcW1, const float* __restrict__ fcb1,
                         const float* __restrict__ fcW2, const float* __restrict__ fcb2,
                         float* __restrict__ out) {
    __shared__ float emb[HIDC + GLOBF];
    __shared__ float wsum[2];
    int b = blockIdx.x, t = threadIdx.x;   // 64 threads
    float c = fmaxf(cnt[b], 1.f);
    emb[t] = pooled[(long)b * HIDC + t] / c;
    if (t < GLOBF) emb[HIDC + t] = glob[b * GLOBF + t];
    __syncthreads();
    float s = fcb1[t];
    for (int j = 0; j < HIDC + GLOBF; ++j) s += emb[j] * fcW1[j * HIDC + t];
    float p = fmaxf(s, 0.f) * fcW2[t];
    for (int off = 16; off; off >>= 1) p += __shfl_xor(p, off, 32);
    if ((t & 31) == 0) wsum[t >> 5] = p;
    __syncthreads();
    if (t == 0) out[b] = wsum[0] + wsum[1] + fcb2[0];
}

// ---------------------------------------------------------------------------
extern "C" void kernel_launch(void* const* d_in, const int* in_sizes, int n_in,
                              void* d_out, int out_size, void* d_ws, size_t ws_size,
                              hipStream_t stream) {
    const float* x    = (const float*)d_in[0];
    const int*   ei   = (const int*)d_in[1];
    const float* ea   = (const float*)d_in[2];
    const int*   bat  = (const int*)d_in[3];
    const float* gf   = (const float*)d_in[4];
    const float *W1 = (const float*)d_in[5],  *As1 = (const float*)d_in[6],  *Ad1 = (const float*)d_in[7];
    const float *We1 = (const float*)d_in[8], *Ae1 = (const float*)d_in[9],  *b1 = (const float*)d_in[10];
    const float *g1 = (const float*)d_in[11], *be1 = (const float*)d_in[12];
    const float *W2 = (const float*)d_in[13], *As2 = (const float*)d_in[14], *Ad2 = (const float*)d_in[15];
    const float *We2 = (const float*)d_in[16], *Ae2 = (const float*)d_in[17], *b2 = (const float*)d_in[18];
    const float *g2 = (const float*)d_in[19], *be2 = (const float*)d_in[20];
    const float *W3 = (const float*)d_in[21], *As3 = (const float*)d_in[22], *Ad3 = (const float*)d_in[23];
    const float *We3 = (const float*)d_in[24], *Ae3 = (const float*)d_in[25], *b3 = (const float*)d_in[26];
    const float *g3 = (const float*)d_in[27], *be3 = (const float*)d_in[28];
    const float *fcW1 = (const float*)d_in[29], *fcb1 = (const float*)d_in[30];
    const float *fcW2 = (const float*)d_in[31], *fcb2 = (const float*)d_in[32];

    // ---- workspace carve-up (~120 MB total; fits global L2) ----
    char* w = (char*)d_ws;
    float* buf0   = (float*)w;                   w += (size_t)N_NODES * HC * 4;
    float* buf1   = (float*)w;                   w += (size_t)N_NODES * HC * 4;
    float* alpha  = (float*)w;                   w += (size_t)ET * HEADS * 4;
    float* a_s    = (float*)w;                   w += (size_t)N_NODES * HEADS * 4;
    float* a_d    = (float*)w;                   w += (size_t)N_NODES * HEADS * 4;
    unsigned* menc = (unsigned*)w;               w += (size_t)N_NODES * HEADS * 4;
    float* den    = (float*)w;                   w += (size_t)N_NODES * HEADS * 4;
    float* sums   = (float*)w;                   w += 64 * 4;
    float* Mproj  = (float*)w;                   w += 64 * 4;
    float* aeself = (float*)w;                   w += 16 * 4;
    float* pooled = (float*)w;                   w += (size_t)NB * HIDC * 4;
    float* cnt    = (float*)w;                   w += (size_t)NB * 4;

    const dim3 Z(2048);
    const int EB = (ET + 255) / 256;             // edge-pass blocks
    const int AGGB = (ET + 7) / 8;               // wave-per-edge blocks
    const int NWB = (N_NODES + 7) / 8;           // wave-per-node blocks
    const int GEMMX = (N_NODES + 127) / 128;

    // edge_attr per-feature mean (for self-loop fill)
    zero_kernel<<<1, 64, 0, stream>>>(sums, 8);
    edge_colsum<<<256, 256, 0, stream>>>(ea, sums);

    // ---------------- layer 1 (KDIM=32, H=4) ----------------
    layer_setup<<<1, 64, 0, stream>>>(We1, Ae1, sums, HEADS, Mproj, aeself);
    gemm_wmma<32><<<dim3(GEMMX, 4), 256, 0, stream>>>(x, W1, buf0, N_NODES, HC);
    node_attn<<<(N_NODES * HEADS + 255) / 256, 256, 0, stream>>>(buf0, As1, Ad1, HEADS, a_s, a_d);
    zero_kernel<<<Z, 256, 0, stream>>>((float*)menc, (long)N_NODES * HEADS);
    zero_kernel<<<Z, 256, 0, stream>>>(den, (long)N_NODES * HEADS);
    zero_kernel<<<Z, 256, 0, stream>>>(buf1, (long)N_NODES * HC);
    edge_alpha<<<EB, 256, 0, stream>>>(ei, ea, Mproj, aeself, a_s, a_d, HEADS, alpha, menc);
    edge_exp<<<EB, 256, 0, stream>>>(ei, HEADS, menc, alpha, den);
    edge_agg<HC><<<AGGB, 256, 0, stream>>>(ei, buf0, alpha, den, HEADS, buf1);
    bias_ln_relu<HC><<<NWB, 256, 0, stream>>>(buf1, b1, g1, be1);      // h1 = buf1

    // ---------------- layer 2 (KDIM=256, H=4) ----------------
    layer_setup<<<1, 64, 0, stream>>>(We2, Ae2, sums, HEADS, Mproj, aeself);
    gemm_wmma<256><<<dim3(GEMMX, 4), 256, 0, stream>>>(buf1, W2, buf0, N_NODES, HC);
    node_attn<<<(N_NODES * HEADS + 255) / 256, 256, 0, stream>>>(buf0, As2, Ad2, HEADS, a_s, a_d);
    zero_kernel<<<Z, 256, 0, stream>>>((float*)menc, (long)N_NODES * HEADS);
    zero_kernel<<<Z, 256, 0, stream>>>(den, (long)N_NODES * HEADS);
    zero_kernel<<<Z, 256, 0, stream>>>(buf1, (long)N_NODES * HC);
    edge_alpha<<<EB, 256, 0, stream>>>(ei, ea, Mproj, aeself, a_s, a_d, HEADS, alpha, menc);
    edge_exp<<<EB, 256, 0, stream>>>(ei, HEADS, menc, alpha, den);
    edge_agg<HC><<<AGGB, 256, 0, stream>>>(ei, buf0, alpha, den, HEADS, buf1);
    bias_ln_relu<HC><<<NWB, 256, 0, stream>>>(buf1, b2, g2, be2);      // h2 = buf1

    // ---------------- layer 3 (KDIM=256 -> 64 cols, H=1) ----------------
    layer_setup<<<1, 64, 0, stream>>>(We3, Ae3, sums, 1, Mproj, aeself);
    gemm_wmma<256><<<dim3(GEMMX, 1), 256, 0, stream>>>(buf1, W3, buf0, N_NODES, HIDC);
    node_attn<<<(N_NODES + 255) / 256, 256, 0, stream>>>(buf0, As3, Ad3, 1, a_s, a_d);
    zero_kernel<<<Z, 256, 0, stream>>>((float*)menc, (long)N_NODES);
    zero_kernel<<<Z, 256, 0, stream>>>(den, (long)N_NODES);
    zero_kernel<<<Z, 256, 0, stream>>>(buf1, (long)N_NODES * HIDC);
    edge_alpha<<<EB, 256, 0, stream>>>(ei, ea, Mproj, aeself, a_s, a_d, 1, alpha, menc);
    edge_exp<<<EB, 256, 0, stream>>>(ei, 1, menc, alpha, den);
    edge_agg<HIDC><<<AGGB, 256, 0, stream>>>(ei, buf0, alpha, den, 1, buf1);
    bias_ln_relu<HIDC><<<NWB, 256, 0, stream>>>(buf1, b3, g3, be3);    // h3 = buf1

    // ---------------- pool + MLP head ----------------
    zero_kernel<<<64, 256, 0, stream>>>(pooled, (long)NB * HIDC + NB); // pooled & cnt contiguous
    pool_accum<<<NWB, 256, 0, stream>>>(buf1, bat, pooled, cnt);
    head_mlp<<<NB, 64, 0, stream>>>(pooled, cnt, gf, fcW1, fcb1, fcW2, fcb2, (float*)d_out);
}
